// MMSingleStreamBlock_56410100466081
// MI455X (gfx1250) — compile-verified, matrix operand order checked by reading, not run on hardware
//
#include <hip/hip_runtime.h>
#include <hip/hip_bf16.h>
#include <math.h>

// ---------------- problem constants ----------------
#define HID     3072
#define NHEADS  24
#define HDIM    128
#define MLPH    12288
#define LTOK    1280
#define IMGL    1024
#define N1      (3*HID + MLPH)     // 21504  (gemm1 N)
#define QKVN    (3*HID)            // 9216
#define CATN    (HID + MLPH)       // 15360  (gemm2 K)
#define EPSV    1e-6f

// ---------------- workspace layout (bytes) ----------------
static constexpr size_t OFF_SILU = 0;
static constexpr size_t OFF_MOD  = OFF_SILU + (size_t)HID * 4;
static constexpr size_t OFF_XMOD = OFF_MOD  + (size_t)3*HID * 4;
static constexpr size_t OFF_QKV  = OFF_XMOD + (size_t)LTOK*HID*2;
static constexpr size_t OFF_Q    = OFF_QKV  + (size_t)LTOK*QKVN*4;
static constexpr size_t OFF_K    = OFF_Q    + (size_t)NHEADS*LTOK*HDIM*2;
static constexpr size_t OFF_V    = OFF_K    + (size_t)NHEADS*LTOK*HDIM*2;
static constexpr size_t OFF_CAT  = OFF_V    + (size_t)NHEADS*LTOK*HDIM*2;

// ---------------- WMMA fragment types ----------------
typedef __attribute__((ext_vector_type(16))) __bf16 v16bf;
typedef __attribute__((ext_vector_type(8)))  __bf16 v8bf;
typedef __attribute__((ext_vector_type(8)))  float  v8f;
typedef __attribute__((ext_vector_type(4)))  int    v4i_t;

// ---------------- CDNA5 async global->LDS copy (ASYNCcnt) ----------------
typedef v4i_t __attribute__((address_space(1)))* as1_b128_t;
typedef v4i_t __attribute__((address_space(3)))* as3_b128_t;

__device__ inline void async_b128(const void* gsrc, void* ldst) {
#if __has_builtin(__builtin_amdgcn_global_load_async_to_lds_b128)
  __builtin_amdgcn_global_load_async_to_lds_b128(
      (as1_b128_t)gsrc, (as3_b128_t)ldst, 0, 0);
#else
  *(int4*)ldst = *(const int4*)gsrc;
#endif
}
__device__ inline void async_wait() {
#if __has_builtin(__builtin_amdgcn_s_wait_asynccnt)
  __builtin_amdgcn_s_wait_asynccnt(0);
#else
  asm volatile("s_wait_asynccnt 0" ::: "memory");
#endif
}

// Load a 16x32 bf16 A-fragment (or 32x16 B-fragment stored n-major) from LDS.
// Per CDNA5 ISA 16-bit matrix layout: lanes 0-15 hold K 0-7 & 16-23,
// lanes 16-31 hold K 8-15 & 24-31 for row/col = lane%16.
__device__ inline v16bf load_frag16(const __bf16* base, int stride, int lane) {
  const int r  = lane & 15;
  const int kb = (lane >> 4) << 3;      // 0 or 8
  union { v16bf v; v8bf h[2]; } u;
  u.h[0] = *(const v8bf*)(base + r * stride + kb);
  u.h[1] = *(const v8bf*)(base + r * stride + kb + 16);
  return u.v;
}

__device__ inline v8f wmma_bf16(v16bf a, v16bf b, v8f c) {
  return __builtin_amdgcn_wmma_f32_16x16x32_bf16(false, a, false, b, (short)0, c, false, false);
}

// gelu(tanh approx) via single fast exp: 0.5v(1+tanh(u)) == v/(1+exp(-2u))
__device__ inline float gelu_fast(float v) {
  float u = 0.7978845608f * (v + 0.044715f * v * v * v);
  return v / (1.0f + __expf(-2.0f * u));
}

// ---------------- kernel 1: silu(vec) ----------------
__global__ void silu_kernel(const float* __restrict__ vec, float* __restrict__ out) {
  int i = blockIdx.x * blockDim.x + threadIdx.x;
  if (i < HID) {
    float v = vec[i];
    out[i] = v / (1.0f + __expf(-v));
  }
}

// ---------------- kernel 2: mod = silu(vec) @ mod_w + mod_b ----------------
__global__ __launch_bounds__(256) void modvec_kernel(const float* __restrict__ s,
                                                     const float* __restrict__ mw,
                                                     const float* __restrict__ mb,
                                                     float* __restrict__ mod) {
  __shared__ float sS[HID];
  const int tid = threadIdx.x;
  for (int i = tid; i < HID; i += 256) sS[i] = s[i];
  __syncthreads();
  const int n = blockIdx.x * 256 + tid;
  float acc = mb[n];
  const float* col = mw + n;
  for (int k = 0; k < HID; k += 8) {
    __builtin_prefetch(col + (size_t)(k + 64) * (3*HID), 0, 1);
    #pragma unroll
    for (int kk = 0; kk < 8; kk++)
      acc += sS[k + kk] * col[(size_t)(k + kk) * (3*HID)];
  }
  mod[n] = acc;
}

// ---------------- shared reduction helpers ----------------
__device__ inline float block_reduce_sum_256(float v, float* red, int tid) {
  red[tid] = v; __syncthreads();
  for (int s = 128; s > 0; s >>= 1) { if (tid < s) red[tid] += red[tid + s]; __syncthreads(); }
  float r = red[0]; __syncthreads();
  return r;
}
__device__ inline float block_reduce_sum_128(float v, float* red, int tid) {
  red[tid] = v; __syncthreads();
  for (int s = 64; s > 0; s >>= 1) { if (tid < s) red[tid] += red[tid + s]; __syncthreads(); }
  float r = red[0]; __syncthreads();
  return r;
}

// ---------------- kernel 3: layernorm + modulate -> bf16 ----------------
__global__ __launch_bounds__(256) void ln_mod_kernel(const float* __restrict__ x,
                                                     const float* __restrict__ mod,
                                                     __bf16* __restrict__ xmod) {
  __shared__ float red[256];
  const int t   = blockIdx.x;
  const int tid = threadIdx.x;
  const float* xr = x + (size_t)t * HID;
  float xs[HID / 256];
  float s1 = 0.f, s2 = 0.f;
  #pragma unroll
  for (int i = 0; i < HID / 256; i++) {
    float v = xr[tid + i * 256];
    xs[i] = v; s1 += v; s2 += v * v;
  }
  s1 = block_reduce_sum_256(s1, red, tid);
  s2 = block_reduce_sum_256(s2, red, tid);
  const float mu   = s1 * (1.0f / HID);
  const float var  = s2 * (1.0f / HID) - mu * mu;
  const float rinv = rsqrtf(var + EPSV);
  __bf16* orow = xmod + (size_t)t * HID;
  #pragma unroll
  for (int i = 0; i < HID / 256; i++) {
    const int n = tid + i * 256;
    float nx = (xs[i] - mu) * rinv;
    float v  = nx * (1.0f + mod[HID + n]) + mod[n];
    orow[n] = (__bf16)v;
  }
}

// ---------------- software-pipelined WMMA GEMM core ----------------
// BM=128 BN=128 BK=32; 8 waves (2x4), wave = 64x32 = 4x2 WMMA tiles.
// A (bf16) streamed via async global->LDS, double buffered.
// W (fp32) register-staged one K-tile ahead, converted to bf16 into LDS.

// ---------------- kernel 4: GEMM1 (x_mod @ w1 + b1) ----------------
__global__ __launch_bounds__(256) void gemm1_kernel(const __bf16* __restrict__ A,
                                                    const float*  __restrict__ W,
                                                    const float*  __restrict__ bias,
                                                    float*  __restrict__ qkv,
                                                    __bf16* __restrict__ cat) {
  __shared__ __bf16 sA[2][128 * 32];
  __shared__ __bf16 sB[2][128 * 32];
  const int tid  = threadIdx.x;
  const int lane = tid & 31;
  const int wave = tid >> 5;
  const int wm   = wave >> 2;
  const int wn   = wave & 3;
  const int m0   = blockIdx.y * 128;
  const int n0   = blockIdx.x * 128;
  const int ar = tid >> 1, ac = (tid & 1) * 16;   // A tile: 16 bf16 / thread
  const int bk = tid >> 3, bn = (tid & 7) * 16;   // W tile: 16 f32 / thread

  float wreg[16];
  auto issueA = [&](int kt, int buf) {
    const __bf16* src = A + (size_t)(m0 + ar) * HID + kt + ac;
    async_b128(src,     &sA[buf][ar * 32 + ac]);
    async_b128(src + 8, &sA[buf][ar * 32 + ac + 8]);
  };
  auto loadB = [&](int kt) {
    const float* src = W + (size_t)(kt + bk) * N1 + n0 + bn;
    __builtin_prefetch(src + (size_t)32 * N1, 0, 1);
    #pragma unroll
    for (int j = 0; j < 16; j += 4) {
      float4 f = *(const float4*)(src + j);
      wreg[j] = f.x; wreg[j+1] = f.y; wreg[j+2] = f.z; wreg[j+3] = f.w;
    }
  };
  auto storeB = [&](int buf) {
    #pragma unroll
    for (int j = 0; j < 16; j++)
      sB[buf][(bn + j) * 32 + bk] = (__bf16)wreg[j];
  };

  v8f acc[4][2];
  #pragma unroll
  for (int i = 0; i < 4; i++)
    #pragma unroll
    for (int j = 0; j < 2; j++)
      #pragma unroll
      for (int e = 0; e < 8; e++) acc[i][j][e] = 0.0f;

  issueA(0, 0);
  loadB(0);
  const int T = HID / 32;
  for (int t = 0; t < T; ++t) {
    const int cur = t & 1, nxt = cur ^ 1;
    storeB(cur);
    async_wait();
    __syncthreads();
    if (t + 1 < T) { issueA((t + 1) * 32, nxt); loadB((t + 1) * 32); }
    v16bf bf[2];
    #pragma unroll
    for (int j = 0; j < 2; j++)
      bf[j] = load_frag16(&sB[cur][(wn * 32 + j * 16) * 32], 32, lane);
    #pragma unroll
    for (int i = 0; i < 4; i++) {
      v16bf af = load_frag16(&sA[cur][(wm * 64 + i * 16) * 32], 32, lane);
      #pragma unroll
      for (int j = 0; j < 2; j++)
        acc[i][j] = wmma_bf16(af, bf[j], acc[i][j]);
    }
    __syncthreads();
  }

  const int cl = lane & 15;
  const int rh = (lane >> 4) * 8;
  #pragma unroll
  for (int i = 0; i < 4; i++)
    #pragma unroll
    for (int j = 0; j < 2; j++) {
      const int n  = n0 + wn * 32 + j * 16 + cl;
      const float bnv = bias[n];
      #pragma unroll
      for (int vj = 0; vj < 8; vj++) {
        const int m = m0 + wm * 64 + i * 16 + vj + rh;
        float v = acc[i][j][vj] + bnv;
        if (n < QKVN) {
          qkv[(size_t)m * QKVN + n] = v;
        } else {
          cat[(size_t)m * CATN + HID + (n - QKVN)] = (__bf16)gelu_fast(v);
        }
      }
    }
}

// ---------------- kernel 5: q/k rmsnorm + RoPE, split to head-major bf16 ----------------
__global__ __launch_bounds__(128) void qknorm_rope_kernel(const float* __restrict__ qkv,
                                                          const float* __restrict__ fcos,
                                                          const float* __restrict__ fsin,
                                                          const float* __restrict__ qn_w,
                                                          const float* __restrict__ kn_w,
                                                          __bf16* __restrict__ Qh,
                                                          __bf16* __restrict__ Kh,
                                                          __bf16* __restrict__ Vh) {
  __shared__ float red[128];
  const int t = blockIdx.x / NHEADS;
  const int h = blockIdx.x % NHEADS;
  const int d = threadIdx.x;
  const float* row = qkv + (size_t)t * QKVN + h * HDIM;
  float q = row[0 * NHEADS * HDIM + d];
  float k = row[1 * NHEADS * HDIM + d];
  float v = row[2 * NHEADS * HDIM + d];

  float qs = block_reduce_sum_128(q * q, red, d);
  float ks = block_reduce_sum_128(k * k, red, d);
  q = q * rsqrtf(qs * (1.0f / HDIM) + EPSV) * qn_w[d];
  k = k * rsqrtf(ks * (1.0f / HDIM) + EPSV) * kn_w[d];

  if (t < IMGL) {
    const float c = fcos[(size_t)t * HDIM + d];
    const float s = fsin[(size_t)t * HDIM + d];
    const float qp = __shfl_xor(q, 1, 32);
    const float kp = __shfl_xor(k, 1, 32);
    const float qr = (d & 1) ? qp : -qp;
    const float kr = (d & 1) ? kp : -kp;
    q = q * c + qr * s;
    k = k * c + kr * s;
  }
  const size_t o = (size_t)h * LTOK * HDIM + (size_t)t * HDIM + d;
  Qh[o] = (__bf16)q;
  Kh[o] = (__bf16)k;
  Vh[o] = (__bf16)v;
}

// ---------------- kernel 6: flash attention (WMMA QK^T and PV) ----------------
__global__ __launch_bounds__(128) void attn_kernel(const __bf16* __restrict__ Qh,
                                                   const __bf16* __restrict__ Kh,
                                                   const __bf16* __restrict__ Vh,
                                                   __bf16* __restrict__ cat) {
  __shared__ __bf16 sQ[64 * HDIM];     // [qrow][d]
  __shared__ __bf16 sK[32 * HDIM];     // [key][d]
  __shared__ __bf16 sVt[HDIM * 32];    // [d][key]
  __shared__ __bf16 sP[4][16 * 32];    // per-wave P tile

  const int tid  = threadIdx.x;
  const int lane = tid & 31;
  const int wave = tid >> 5;
  const int head = blockIdx.x / (LTOK / 64);
  const int q0   = (blockIdx.x % (LTOK / 64)) * 64;
  const __bf16* Qb = Qh + (size_t)head * LTOK * HDIM + (size_t)q0 * HDIM;
  const __bf16* Kb = Kh + (size_t)head * LTOK * HDIM;
  const __bf16* Vb = Vh + (size_t)head * LTOK * HDIM;
  const float scale = 0.08838834764831845f;   // 1/sqrt(128)

  { // 64x128 Q tile via async global->LDS (32 bf16 / thread, 8-elem chunks 16 apart)
    const int r = tid >> 1, c = (tid & 1) * 64;
    #pragma unroll
    for (int j = 0; j < 4; j++)
      async_b128(Qb + (size_t)r * HDIM + c + j * 16, &sQ[r * HDIM + c + j * 16]);
  }
  async_wait();
  __syncthreads();

  v16bf qf[4];
  #pragma unroll
  for (int s = 0; s < 4; s++)
    qf[s] = load_frag16(&sQ[(wave * 16) * HDIM + s * 32], HDIM, lane);

  v8f O[8];
  float mrun[8], lrun[8];
  #pragma unroll
  for (int i = 0; i < 8; i++) {
    mrun[i] = -3.0e38f; lrun[i] = 0.f;
    #pragma unroll
    for (int e = 0; e < 8; e++) O[i][e] = 0.f;
  }

  for (int kb = 0; kb < LTOK; kb += 32) {
    __syncthreads();   // previous iteration's reads done
    { // K tile via async copy (32 elems / thread, contiguous)
      const __bf16* src = Kb + (size_t)kb * HDIM + tid * 32;
      #pragma unroll
      for (int j = 0; j < 4; j++)
        async_b128(src + j * 8, &sK[tid * 32 + j * 8]);
    }
    { // V tile transposed: vector global reads, scatter bf16 to LDS
      const int key = tid >> 2, db = (tid & 3) * 32;
      const __bf16* vs = Vb + (size_t)(kb + key) * HDIM + db;
      union { int4 q[4]; __bf16 e[32]; } vv;
      #pragma unroll
      for (int j = 0; j < 4; j++) vv.q[j] = *(const int4*)(vs + j * 8);
      #pragma unroll
      for (int j = 0; j < 32; j++) sVt[(db + j) * 32 + key] = vv.e[j];
    }
    async_wait();
    __syncthreads();

    v8f s0, s1;
    #pragma unroll
    for (int e = 0; e < 8; e++) { s0[e] = 0.f; s1[e] = 0.f; }
    #pragma unroll
    for (int s = 0; s < 4; s++) {
      v16bf b0 = load_frag16(&sK[(0 * 16) * HDIM + s * 32], HDIM, lane);
      v16bf b1 = load_frag16(&sK[(1 * 16) * HDIM + s * 32], HDIM, lane);
      s0 = wmma_bf16(qf[s], b0, s0);
      s1 = wmma_bf16(qf[s], b1, s1);
    }

    const int cl = lane & 15;
    const int rh = (lane >> 4) * 8;
    #pragma unroll
    for (int vj = 0; vj < 8; vj++) {
      float a0 = s0[vj] * scale, a1 = s1[vj] * scale;
      float mx = fmaxf(a0, a1);
      #pragma unroll
      for (int m = 1; m < 16; m <<= 1) mx = fmaxf(mx, __shfl_xor(mx, m, 32));
      const float mnew = fmaxf(mrun[vj], mx);
      const float corr = __expf(mrun[vj] - mnew);
      const float p0 = __expf(a0 - mnew);
      const float p1 = __expf(a1 - mnew);
      float rs = p0 + p1;
      #pragma unroll
      for (int m = 1; m < 16; m <<= 1) rs += __shfl_xor(rs, m, 32);
      lrun[vj] = lrun[vj] * corr + rs;
      mrun[vj] = mnew;
      #pragma unroll
      for (int dt = 0; dt < 8; dt++) O[dt][vj] *= corr;
      const int r = vj + rh;
      sP[wave][r * 32 + cl]      = (__bf16)p0;
      sP[wave][r * 32 + 16 + cl] = (__bf16)p1;
    }

    v16bf pf = load_frag16(&sP[wave][0], 32, lane);
    #pragma unroll
    for (int dt = 0; dt < 8; dt++) {
      v16bf bv = load_frag16(&sVt[(dt * 16) * 32], 32, lane);
      O[dt] = wmma_bf16(pf, bv, O[dt]);
    }
  }

  const int cl = lane & 15;
  const int rh = (lane >> 4) * 8;
  #pragma unroll
  for (int vj = 0; vj < 8; vj++) {
    const float rl = 1.0f / lrun[vj];
    const int m = q0 + wave * 16 + vj + rh;
    #pragma unroll
    for (int dt = 0; dt < 8; dt++)
      cat[(size_t)m * CATN + head * HDIM + dt * 16 + cl] = (__bf16)(O[dt][vj] * rl);
  }
}

// ---------------- kernel 7: GEMM2 (concat @ w2 + b2), gate + residual ----------------
__global__ __launch_bounds__(256) void gemm2_kernel(const __bf16* __restrict__ A,
                                                    const float*  __restrict__ W,
                                                    const float*  __restrict__ bias,
                                                    const float*  __restrict__ x,
                                                    const float*  __restrict__ mod,
                                                    float* __restrict__ out) {
  __shared__ __bf16 sA[2][128 * 32];
  __shared__ __bf16 sB[2][128 * 32];
  const int tid  = threadIdx.x;
  const int lane = tid & 31;
  const int wave = tid >> 5;
  const int wm   = wave >> 2;
  const int wn   = wave & 3;
  const int m0   = blockIdx.y * 128;
  const int n0   = blockIdx.x * 128;
  const int ar = tid >> 1, ac = (tid & 1) * 16;
  const int bk = tid >> 3, bn = (tid & 7) * 16;

  float wreg[16];
  auto issueA = [&](int kt, int buf) {
    const __bf16* src = A + (size_t)(m0 + ar) * CATN + kt + ac;
    async_b128(src,     &sA[buf][ar * 32 + ac]);
    async_b128(src + 8, &sA[buf][ar * 32 + ac + 8]);
  };
  auto loadB = [&](int kt) {
    const float* src = W + (size_t)(kt + bk) * HID + n0 + bn;
    __builtin_prefetch(src + (size_t)32 * HID, 0, 1);
    #pragma unroll
    for (int j = 0; j < 16; j += 4) {
      float4 f = *(const float4*)(src + j);
      wreg[j] = f.x; wreg[j+1] = f.y; wreg[j+2] = f.z; wreg[j+3] = f.w;
    }
  };
  auto storeB = [&](int buf) {
    #pragma unroll
    for (int j = 0; j < 16; j++)
      sB[buf][(bn + j) * 32 + bk] = (__bf16)wreg[j];
  };

  v8f acc[4][2];
  #pragma unroll
  for (int i = 0; i < 4; i++)
    #pragma unroll
    for (int j = 0; j < 2; j++)
      #pragma unroll
      for (int e = 0; e < 8; e++) acc[i][j][e] = 0.0f;

  issueA(0, 0);
  loadB(0);
  const int T = CATN / 32;
  for (int t = 0; t < T; ++t) {
    const int cur = t & 1, nxt = cur ^ 1;
    storeB(cur);
    async_wait();
    __syncthreads();
    if (t + 1 < T) { issueA((t + 1) * 32, nxt); loadB((t + 1) * 32); }
    v16bf bf[2];
    #pragma unroll
    for (int j = 0; j < 2; j++)
      bf[j] = load_frag16(&sB[cur][(wn * 32 + j * 16) * 32], 32, lane);
    #pragma unroll
    for (int i = 0; i < 4; i++) {
      v16bf af = load_frag16(&sA[cur][(wm * 64 + i * 16) * 32], 32, lane);
      #pragma unroll
      for (int j = 0; j < 2; j++)
        acc[i][j] = wmma_bf16(af, bf[j], acc[i][j]);
    }
    __syncthreads();
  }

  const int cl = lane & 15;
  const int rh = (lane >> 4) * 8;
  const float* gate = mod + 2 * HID;
  #pragma unroll
  for (int i = 0; i < 4; i++)
    #pragma unroll
    for (int j = 0; j < 2; j++) {
      const int n = n0 + wn * 32 + j * 16 + cl;
      const float bnv = bias[n];
      const float g   = gate[n];
      #pragma unroll
      for (int vj = 0; vj < 8; vj++) {
        const int m = m0 + wm * 64 + i * 16 + vj + rh;
        out[(size_t)m * HID + n] = x[(size_t)m * HID + n] + (acc[i][j][vj] + bnv) * g;
      }
    }
}

// ---------------- host orchestration ----------------
extern "C" void kernel_launch(void* const* d_in, const int* in_sizes, int n_in,
                              void* d_out, int out_size, void* d_ws, size_t ws_size,
                              hipStream_t stream) {
  (void)in_sizes; (void)n_in; (void)out_size; (void)ws_size;
  const float* x     = (const float*)d_in[0];
  const float* vec   = (const float*)d_in[1];
  const float* fcos  = (const float*)d_in[2];
  const float* fsin  = (const float*)d_in[3];
  const float* mod_w = (const float*)d_in[4];
  const float* mod_b = (const float*)d_in[5];
  const float* w1    = (const float*)d_in[6];
  const float* b1    = (const float*)d_in[7];
  const float* w2    = (const float*)d_in[8];
  const float* b2    = (const float*)d_in[9];
  const float* qn_w  = (const float*)d_in[10];
  const float* kn_w  = (const float*)d_in[11];

  char* ws = (char*)d_ws;
  float*  silu_b = (float*)(ws + OFF_SILU);
  float*  mod    = (float*)(ws + OFF_MOD);
  __bf16* xmod   = (__bf16*)(ws + OFF_XMOD);
  float*  qkv    = (float*)(ws + OFF_QKV);
  __bf16* Qh     = (__bf16*)(ws + OFF_Q);
  __bf16* Kh     = (__bf16*)(ws + OFF_K);
  __bf16* Vh     = (__bf16*)(ws + OFF_V);
  __bf16* cat    = (__bf16*)(ws + OFF_CAT);
  float*  out    = (float*)d_out;

  silu_kernel<<<(HID + 255) / 256, 256, 0, stream>>>(vec, silu_b);
  modvec_kernel<<<(3 * HID) / 256, 256, 0, stream>>>(silu_b, mod_w, mod_b, mod);
  ln_mod_kernel<<<LTOK, 256, 0, stream>>>(x, mod, xmod);
  gemm1_kernel<<<dim3(N1 / 128, LTOK / 128), 256, 0, stream>>>(xmod, w1, b1, qkv, cat);
  qknorm_rope_kernel<<<LTOK * NHEADS, 128, 0, stream>>>(qkv, fcos, fsin, qn_w, kn_w, Qh, Kh, Vh);
  attn_kernel<<<NHEADS * (LTOK / 64), 128, 0, stream>>>(Qh, Kh, Vh, cat);
  gemm2_kernel<<<dim3(HID / 128, LTOK / 128), 256, 0, stream>>>(cat, w2, b2, x, mod, out);
}
